// LocalAttentionBlock_48103633715277
// MI455X (gfx1250) — matrix operand order repaired and checked by reading, not compiled, not run on hardware
//
#include <hip/hip_runtime.h>

// ---------------------------------------------------------------------------
// LocalAttentionBlock for MI455X (gfx1250): bf16 WMMA everywhere, f32 accum.
// B=4, T=2048, WIDTH=2048, NUM_HEADS=16, HEAD_DIM=128, WINDOW=1024.
// GEMMs are LDS-staged (double buffered) using async global->LDS copies
// (GLOBAL_LOAD_ASYNC_TO_LDS_B128, ASYNCcnt) when available.
// ---------------------------------------------------------------------------

typedef __bf16 bf16;
typedef __attribute__((ext_vector_type(16))) __bf16 v16bf;
typedef __attribute__((ext_vector_type(4)))  __bf16 v4bf;
typedef __attribute__((ext_vector_type(8)))  float  v8f;
typedef __attribute__((ext_vector_type(4)))  float  v4f;
typedef __attribute__((ext_vector_type(4)))  int    v4i;
typedef __attribute__((ext_vector_type(4)))  unsigned int uint4v;

#define T_SEQ   2048
#define WIDTHC  2048
#define HEADS   16
#define HDIM    128
#define WINDOW  1024
#define BATCH   4
#define ROWS    (BATCH * T_SEQ)          // 8192
#define QSCALE  0.08838834764831845f     // 1/sqrt(128)
#define LN10000_OVER_32 0.28782313662425574f

#define AS1 __attribute__((address_space(1)))
#define AS3 __attribute__((address_space(3)))

#if defined(__has_builtin)
#if __has_builtin(__builtin_amdgcn_global_load_async_to_lds_b128)
#define HAVE_ASYNC 1
#endif
#endif
#ifndef HAVE_ASYNC
#define HAVE_ASYNC 0
#endif

#if HAVE_ASYNC
#if __has_builtin(__builtin_amdgcn_s_wait_asynccnt)
#define WAIT_ASYNC(n) __builtin_amdgcn_s_wait_asynccnt(n)
#else
#define WAIT_ASYNC(n) asm volatile("s_wait_asynccnt %0" ::"i"(n) : "memory")
#endif
#endif

union FragU { v16bf v; uint4v q[2]; };

__device__ __forceinline__ v8f zero8() {
  v8f z;
#pragma unroll
  for (int i = 0; i < 8; ++i) z[i] = 0.0f;
  return z;
}

// ---- global fragment loaders (used by attention) ---------------------------
// A fragment (16x32 bf16, M x K): lane l -> row (l&15);
// lanes 0-15 hold K in {0..7, 16..23}, lanes 16-31 hold K in {8..15, 24..31}.
__device__ __forceinline__ v16bf load_fragA(const bf16* __restrict__ base,
                                            long ld, int row0, int k0) {
  const int l  = threadIdx.x & 31;
  const int hi = l >> 4;
  const bf16* p = base + (long)(row0 + (l & 15)) * ld + (k0 + hi * 8);
  FragU u;
  u.q[0] = *(const uint4v*)(p);
  u.q[1] = *(const uint4v*)(p + 16);
  return u.v;
}

// B fragment (32x16 bf16, K x N) from row-major [N][K] operand:
// lane l -> column (l&15); lanes 0-15 hold K in {0..7, 16..23}, lanes 16-31 hold K in {8..15, 24..31}.
__device__ __forceinline__ v16bf load_fragB(const bf16* __restrict__ base,
                                            long ld, int col0, int k0) {
  const int l = threadIdx.x & 31;
  const bf16* p = base + (long)(col0 + (l & 15)) * ld + (k0 + (l >> 4) * 8);
  FragU u;
  u.q[0] = *(const uint4v*)(p);
  u.q[1] = *(const uint4v*)(p + 16);
  return u.v;
}

// ---- LDS fragment loaders (tiles staged row-major, padded stride) ----------
#define LDS_STRIDE 40   // 32 elems + 8 pad -> 80B rows (16B aligned, bank-spread)

__device__ __forceinline__ v16bf lds_fragA(const bf16* base, int row0) {
  const int l  = threadIdx.x & 31;
  const int hi = l >> 4;
  const bf16* p = base + (row0 + (l & 15)) * LDS_STRIDE + hi * 8;
  FragU u;
  u.q[0] = *(const uint4v*)(p);
  u.q[1] = *(const uint4v*)(p + 16);
  return u.v;
}

__device__ __forceinline__ v16bf lds_fragB(const bf16* base, int row0) {
  const int l = threadIdx.x & 31;
  const bf16* p = base + (row0 + (l & 15)) * LDS_STRIDE + (l >> 4) * 8;
  FragU u;
  u.q[0] = *(const uint4v*)(p);
  u.q[1] = *(const uint4v*)(p + 16);
  return u.v;
}

__device__ __forceinline__ v8f wmma_bf16(v16bf a, v16bf b, v8f c) {
  return __builtin_amdgcn_wmma_f32_16x16x32_bf16(false, a, false, b,
                                                 (short)0, c, false, false);
}

// one 16B chunk global -> LDS (async if available, else synchronous)
__device__ __forceinline__ void copy16(const bf16* g, bf16* l) {
#if HAVE_ASYNC
  __builtin_amdgcn_global_load_async_to_lds_b128((AS1 v4i*)g, (AS3 v4i*)l, 0, 0);
#else
  *(uint4v*)l = *(const uint4v*)g;
#endif
}

// ---------------------------------------------------------------------------
// f32 -> bf16 cast, 4 elements / thread
// ---------------------------------------------------------------------------
__global__ __launch_bounds__(256)
void cast_f32_to_bf16(const float* __restrict__ in, bf16* __restrict__ out,
                      long n) {
  long i = ((long)blockIdx.x * 256 + threadIdx.x) * 4;
  if (i >= n) return;
  v4f xv = *(const v4f*)(in + i);
  v4bf y;
#pragma unroll
  for (int j = 0; j < 4; ++j) y[j] = (bf16)xv[j];
  *(v4bf*)(out + i) = y;
}

// ---------------------------------------------------------------------------
// C = A[M,K] * W[N,K]^T, LDS-staged, double buffered.
// Block tile 128x128xBK32, 8 waves -> each wave 32(M) x 64(N) = 2x4 frags.
// ROPE  : rotary embedding fused into the epilogue (pair (j,j+32) is the
//         same lane, two fragments apart, for a 64-aligned wave span).
// OUTF32: f32 store + bias (final projection).
// TRANSV: store transposed as vt[b][col][t] (V projection, N==128).
// ---------------------------------------------------------------------------
template<bool ROPE, bool OUTF32, bool TRANSV>
__global__ __launch_bounds__(256)
void gemm_wmma_bf16(const bf16* __restrict__ A, const bf16* __restrict__ W,
                    void* __restrict__ Cout, int M, int N, int K,
                    const int* __restrict__ pos, const float* __restrict__ bias,
                    float scale) {
  __shared__ __align__(16) bf16 As[2][128 * LDS_STRIDE];
  __shared__ __align__(16) bf16 Bs[2][128 * LDS_STRIDE];

  const int tid  = threadIdx.x;
  const int l    = tid & 31;
  const int wave = tid >> 5;
  const int wm   = wave >> 1;        // 0..3
  const int wn   = wave & 1;         // 0..1
  const int m0b  = blockIdx.x * 128;
  const int n0b  = blockIdx.y * 128;
  const int lr   = l & 15;
  const int g    = l >> 4;

  // stage one 128x32 A tile + 128x32 B tile into LDS buffer db.
  // 512 chunks of 16B each per tile; 256 threads x 2 chunks.
  auto stage = [&](int k0, int db) {
#pragma unroll
    for (int c = tid; c < 512; c += 256) {
      const int row  = c >> 2;
      const int part = (c & 3) * 8;                  // 0,8,16,24
      copy16(A + (long)(m0b + row) * K + k0 + part,
             &As[db][row * LDS_STRIDE + part]);
      copy16(W + (long)(n0b + row) * K + k0 + part,
             &Bs[db][row * LDS_STRIDE + part]);
    }
  };

  v8f acc[2][4];
#pragma unroll
  for (int i = 0; i < 2; ++i)
#pragma unroll
    for (int j = 0; j < 4; ++j) acc[i][j] = zero8();

  stage(0, 0);
  for (int k0 = 0; k0 < K; k0 += 32) {
    const int db = (k0 >> 5) & 1;
    if (k0 + 32 < K) {
      stage(k0 + 32, db ^ 1);
#if HAVE_ASYNC
      WAIT_ASYNC(4);                 // current stage's 4 copies have landed
#endif
    } else {
#if HAVE_ASYNC
      WAIT_ASYNC(0);
#endif
    }
    __syncthreads();                 // everyone's stage-k data visible

    v16bf a0 = lds_fragA(As[db], wm * 32);
    v16bf a1 = lds_fragA(As[db], wm * 32 + 16);
#pragma unroll
    for (int j = 0; j < 4; ++j) {
      v16bf b = lds_fragB(Bs[db], wn * 64 + j * 16);
      acc[0][j] = wmma_bf16(a0, b, acc[0][j]);
      acc[1][j] = wmma_bf16(a1, b, acc[1][j]);
    }
    __syncthreads();                 // done reading db before it is re-staged
  }

  const int m0 = m0b + wm * 32;
  const int n0 = n0b + wn * 64;

  if (ROPE && ((n0 & 127) == 0)) {   // rope half of the head
#pragma unroll
    for (int mi = 0; mi < 2; ++mi) {
#pragma unroll
      for (int r = 0; r < 8; ++r) {
        int row = m0 + mi * 16 + g * 8 + r;
        float p = (float)pos[row & (T_SEQ - 1)];
#pragma unroll
        for (int f = 0; f < 2; ++f) {
          float jj   = (float)(f * 16 + lr);              // 0..31
          float invf = __expf(jj * -LN10000_OVER_32);     // 10000^(-j/32)
          float ang  = p * invf;
          float sn = __sinf(ang), cs = __cosf(ang);
          float x1 = acc[mi][f][r], x2 = acc[mi][f + 2][r];
          acc[mi][f][r]     = x1 * cs - x2 * sn;
          acc[mi][f + 2][r] = x2 * cs + x1 * sn;
        }
      }
    }
  }

#pragma unroll
  for (int mi = 0; mi < 2; ++mi) {
#pragma unroll
    for (int j = 0; j < 4; ++j) {
#pragma unroll
      for (int r = 0; r < 8; ++r) {
        int row = m0 + mi * 16 + g * 8 + r;
        int col = n0 + j * 16 + lr;
        float v = acc[mi][j][r];
        if (OUTF32) {
          ((float*)Cout)[(long)row * N + col] = v + bias[col];
        } else if (TRANSV) {
          long idx = ((long)(row >> 11) * HDIM + col) * T_SEQ + (row & (T_SEQ - 1));
          ((bf16*)Cout)[idx] = (bf16)(v * scale);
        } else {
          ((bf16*)Cout)[(long)row * N + col] = (bf16)(v * scale);
        }
      }
    }
  }
}

// ---------------------------------------------------------------------------
// Flash-style windowed MQA attention. One wave per (b, head, 16-query tile).
// q:  bf16 [b*T][2048] (RoPE'd, pre-scaled) | k: bf16 [b*T][128] (RoPE'd)
// vt: bf16 [b][128][T] (transposed so PV B-frags load contiguously)
// ---------------------------------------------------------------------------
__global__ __launch_bounds__(32)
void attn_wmma(const bf16* __restrict__ qb, const bf16* __restrict__ kb,
               const bf16* __restrict__ vtb, bf16* __restrict__ encb) {
  const int l  = threadIdx.x & 31;
  const int lr = l & 15;
  const int g  = l >> 4;
  const int qt = blockIdx.x & 127;
  const int hn = (blockIdx.x >> 7) & 15;
  const int b  = blockIdx.x >> 11;
  const int t0 = qt * 16;

  __shared__ __align__(16) bf16 psh[16][32];

  const bf16* qbase = qb + ((long)(b * T_SEQ) + t0) * WIDTHC + hn * HDIM;
  v16bf qa[4];
#pragma unroll
  for (int kk = 0; kk < 4; ++kk) qa[kk] = load_fragA(qbase, WIDTHC, 0, kk * 32);

  v8f o[8];
#pragma unroll
  for (int f = 0; f < 8; ++f) o[f] = zero8();
  float mx[8], sm[8];
#pragma unroll
  for (int r = 0; r < 8; ++r) { mx[r] = -3.0e38f; sm[r] = 0.0f; }

  int lo = t0 - WINDOW;
  if (lo < 0) lo = 0;
  lo &= ~31;

  for (int s0 = lo; s0 <= t0 + 15; s0 += 32) {
    v8f Lg[2];
#pragma unroll
    for (int j = 0; j < 2; ++j) {
      Lg[j] = zero8();
      const bf16* kbase = kb + (long)(b * T_SEQ + s0 + j * 16) * HDIM;
#pragma unroll
      for (int kk = 0; kk < 4; ++kk) {
        v16bf kf = load_fragB(kbase, HDIM, 0, kk * 32);
        Lg[j] = wmma_bf16(qa[kk], kf, Lg[j]);
      }
    }
#pragma unroll
    for (int r = 0; r < 8; ++r) {
      int t = t0 + g * 8 + r;
#pragma unroll
      for (int j = 0; j < 2; ++j) {
        int s = s0 + j * 16 + lr;
        if (!((s <= t) && (t <= s + WINDOW))) Lg[j][r] = -3.0e38f;
      }
    }
#pragma unroll
    for (int r = 0; r < 8; ++r) {
      float v = fmaxf(Lg[0][r], Lg[1][r]);
#pragma unroll
      for (int d = 1; d < 16; d <<= 1) v = fmaxf(v, __shfl_xor(v, d, 32));
      float nm   = fmaxf(mx[r], v);
      float corr = __expf(mx[r] - nm);
      mx[r] = nm;
      sm[r] *= corr;
#pragma unroll
      for (int f = 0; f < 8; ++f) o[f][r] *= corr;
      float p0 = __expf(Lg[0][r] - nm);
      float p1 = __expf(Lg[1][r] - nm);
      Lg[0][r] = p0; Lg[1][r] = p1;
      float ps = p0 + p1;
#pragma unroll
      for (int d = 1; d < 16; d <<= 1) ps += __shfl_xor(ps, d, 32);
      sm[r] += ps;
    }
#pragma unroll
    for (int r = 0; r < 8; ++r) {
      psh[g * 8 + r][lr]      = (bf16)Lg[0][r];
      psh[g * 8 + r][16 + lr] = (bf16)Lg[1][r];
    }
    __syncthreads();
    FragU u;
    u.q[0] = *(const uint4v*)&psh[lr][g * 8];
    u.q[1] = *(const uint4v*)&psh[lr][16 + g * 8];
    v16bf pa = u.v;
    __syncthreads();
#pragma unroll
    for (int f = 0; f < 8; ++f) {
      const bf16* vbase = vtb + ((long)b * HDIM + f * 16) * T_SEQ;
      v16bf vf = load_fragB(vbase, T_SEQ, 0, s0);
      o[f] = wmma_bf16(pa, vf, o[f]);
    }
  }

#pragma unroll
  for (int r = 0; r < 8; ++r) {
    float inv = 1.0f / sm[r];
    long row = (long)b * T_SEQ + t0 + g * 8 + r;
#pragma unroll
    for (int f = 0; f < 8; ++f) {
      encb[row * WIDTHC + hn * HDIM + f * 16 + lr] = (bf16)(o[f][r] * inv);
    }
  }
}

// ---------------------------------------------------------------------------
extern "C" void kernel_launch(void* const* d_in, const int* in_sizes, int n_in,
                              void* d_out, int out_size, void* d_ws,
                              size_t ws_size, hipStream_t stream) {
  const float* x    = (const float*)d_in[0];
  const int*   spos = (const int*)d_in[1];
  // d_in[2] = attention_mask (recomputed arithmetically; unused)
  const float* wq = (const float*)d_in[3];
  const float* wk = (const float*)d_in[4];
  const float* wv = (const float*)d_in[5];
  const float* wo = (const float*)d_in[6];
  const float* bo = (const float*)d_in[7];
  float* out = (float*)d_out;

  char* ws = (char*)d_ws;
  auto take = [&](long bytes) { char* p = ws; ws += bytes; return p; };
  bf16* xb   = (bf16*)take((long)ROWS * WIDTHC * 2);
  bf16* wqb  = (bf16*)take((long)WIDTHC * WIDTHC * 2);
  bf16* wkb  = (bf16*)take((long)HDIM * WIDTHC * 2);
  bf16* wvb  = (bf16*)take((long)HDIM * WIDTHC * 2);
  bf16* wob  = (bf16*)take((long)WIDTHC * WIDTHC * 2);
  bf16* qbuf = (bf16*)take((long)ROWS * WIDTHC * 2);
  bf16* kbuf = (bf16*)take((long)ROWS * HDIM * 2);
  bf16* vtb  = (bf16*)take((long)ROWS * HDIM * 2);
  bf16* encb = (bf16*)take((long)ROWS * WIDTHC * 2);

  auto cast = [&](const float* src, bf16* dst, long n) {
    cast_f32_to_bf16<<<(unsigned)(n / 1024), 256, 0, stream>>>(src, dst, n);
  };
  cast(x,  xb,  (long)ROWS * WIDTHC);
  cast(wq, wqb, (long)WIDTHC * WIDTHC);
  cast(wk, wkb, (long)HDIM * WIDTHC);
  cast(wv, wvb, (long)HDIM * WIDTHC);
  cast(wo, wob, (long)WIDTHC * WIDTHC);

  dim3 blk(256);
  // Q = rope(x @ wq^T) * 1/sqrt(128)
  gemm_wmma_bf16<true, false, false><<<dim3(ROWS / 128, WIDTHC / 128), blk, 0, stream>>>(
      xb, wqb, qbuf, ROWS, WIDTHC, WIDTHC, spos, nullptr, QSCALE);
  // K = rope(x @ wk^T)
  gemm_wmma_bf16<true, false, false><<<dim3(ROWS / 128, 1), blk, 0, stream>>>(
      xb, wkb, kbuf, ROWS, HDIM, WIDTHC, spos, nullptr, 1.0f);
  // V^T = (x @ wv^T)^T
  gemm_wmma_bf16<false, false, true><<<dim3(ROWS / 128, 1), blk, 0, stream>>>(
      xb, wvb, vtb, ROWS, HDIM, WIDTHC, spos, nullptr, 1.0f);
  // windowed flash attention
  attn_wmma<<<BATCH * HEADS * (T_SEQ / 16), 32, 0, stream>>>(qbuf, kbuf, vtb, encb);
  // out = enc @ wo^T + b_out  (f32)
  gemm_wmma_bf16<false, true, false><<<dim3(ROWS / 128, WIDTHC / 128), blk, 0, stream>>>(
      encb, wob, out, ROWS, WIDTHC, WIDTHC, nullptr, bo, 1.0f);
}
